// GCN_58643483460179
// MI455X (gfx1250) — matrix operand order repaired and checked by reading, not compiled, not run on hardware
//
#include <hip/hip_runtime.h>
#include <stdint.h>

#define BN_   100000   // B*N nodes
#define ET_   800000   // B*E edges
#define Hdim  512
#define OUTD  256
#define Bg    4
#define Nn    25000
#define LH    12
#define NCONV 3

typedef __attribute__((ext_vector_type(16))) __bf16          v16bf;
typedef __attribute__((ext_vector_type(8)))  float           v8f;
typedef __attribute__((ext_vector_type(8)))  unsigned short  ushort8;
typedef __attribute__((ext_vector_type(16))) unsigned short  ushort16;

__device__ __forceinline__ unsigned short f2bf(float f) {
  unsigned int u = __float_as_uint(f);
  u += 0x7FFFu + ((u >> 16) & 1u);          // round-to-nearest-even
  return (unsigned short)(u >> 16);
}
__device__ __forceinline__ float lrelu(float v) { return v > 0.f ? v : 0.01f * v; }

__global__ void fill_f32(float* __restrict__ p, float v, int n) {
  int i = blockIdx.x * blockDim.x + threadIdx.x;
  if (i < n) p[i] = v;
}

__global__ void edge_deg(const int* __restrict__ dst, float* __restrict__ deg) {
  int e = blockIdx.x * blockDim.x + threadIdx.x;
  if (e < ET_) atomicAdd(&deg[dst[e]], 1.0f);
}

__global__ void make_dinv(const float* __restrict__ deg, float* __restrict__ dinv) {
  int i = blockIdx.x * blockDim.x + threadIdx.x;
  if (i < BN_) dinv[i] = rsqrtf(deg[i]);
}

// layer-1 aggregation is rank-1: only a scalar per node needed
__global__ void edge_scalar(const int* __restrict__ src, const int* __restrict__ dst,
                            const float* __restrict__ x, const float* __restrict__ dinv,
                            float* __restrict__ sagg) {
  int e = blockIdx.x * blockDim.x + threadIdx.x;
  if (e >= ET_) return;
  int s = src[e], d = dst[e];
  atomicAdd(&sagg[d], x[s] * dinv[s] * dinv[d]);
}

__global__ void node_layer1(const float* __restrict__ x, const float* __restrict__ dinv,
                            const float* __restrict__ sagg, const float* __restrict__ W1,
                            const float* __restrict__ b1,
                            float* __restrict__ h, unsigned short* __restrict__ hb) {
  int idx = blockIdx.x * blockDim.x + threadIdx.x;
  if (idx >= BN_ * Hdim) return;
  int i = idx >> 9;          // / 512
  int c = idx & 511;
  float di = dinv[i];
  float xi = x[i];
  float conv = (sagg[i] + xi * di * di) * W1[c] + b1[c];
  float v = xi + lrelu(conv);
  h[idx]  = v;
  hb[idx] = f2bf(v);
}

// Pack W [K=512][N=512] f32 -> WMMA B-fragment-contiguous bf16:
// Wp[((kt*32 + nt)*32 + lane)*16 + e], K = kt*32 + 8*(lane>=16) + (e&7) + 16*(e>=8), N = nt*16 + lane%16
__global__ void pack_w(const float* __restrict__ W, unsigned short* __restrict__ Wp) {
  int idx = blockIdx.x * blockDim.x + threadIdx.x;
  const int total = (Hdim / 32) * (Hdim / 16) * 32 * 16;   // 262144
  if (idx >= total) return;
  int e    = idx & 15;
  int lane = (idx >> 4) & 31;
  int tile = idx >> 9;
  int nt   = tile % (Hdim / 16);
  int kt   = tile / (Hdim / 16);
  int k = kt * 32 + ((lane >> 4) << 3) + (e & 7) + ((e >> 3) << 4);
  int n = nt * 16 + (lane & 15);
  Wp[idx] = f2bf(W[k * Hdim + n]);
}

// t[100000x512] = hb[100000x512](bf16) @ Wp(packed bf16), f32 accumulate.
// One wave -> 32x32 output tile (2x2 WMMA accumulators).
__global__ __launch_bounds__(256) void gemm_bf16(const unsigned short* __restrict__ hb,
                                                 const unsigned short* __restrict__ Wp,
                                                 float* __restrict__ t) {
  const int lane = threadIdx.x & 31;
  const int wid  = blockIdx.x * 8 + (threadIdx.x >> 5);
  const int NT2  = Hdim / 32;                 // 16
  const int mt2  = wid / NT2;
  const int nt2  = wid - mt2 * NT2;
  if (mt2 >= BN_ / 32) return;                // wave-uniform guard: EXEC all-ones below
  const int mbase   = mt2 * 32;
  const int nbase   = nt2 * 32;
  const int halfsel = lane >> 4;
  const int mrow    = lane & 15;

  v8f acc[2][2] = {};
  for (int kb = 0; kb < Hdim / 32; ++kb) {
    const int kbase = kb * 32;
    v16bf a[2], bf[2];
    #pragma unroll
    for (int mi = 0; mi < 2; ++mi) {
      // A fragment: lanes 0-15 take K chunks [k..k+8) & [k+16..k+24); lanes 16-31 shifted +8
      const unsigned short* row =
          hb + (size_t)(mbase + mi * 16 + mrow) * Hdim + kbase + halfsel * 8;
      ushort8 lo = *(const ushort8*)(row);
      ushort8 hi = *(const ushort8*)(row + 16);
      ushort16 u;
      #pragma unroll
      for (int q = 0; q < 8; ++q) { u[q] = lo[q]; u[q + 8] = hi[q]; }
      a[mi] = __builtin_bit_cast(v16bf, u);
    }
    #pragma unroll
    for (int ni = 0; ni < 2; ++ni) {
      const int nt = nt2 * 2 + ni;
      const unsigned short* wp =
          Wp + (((size_t)kb * (Hdim / 16) + nt) * 32 + lane) * 16;
      ushort8 lo = *(const ushort8*)(wp);
      ushort8 hi = *(const ushort8*)(wp + 8);
      ushort16 u;
      #pragma unroll
      for (int q = 0; q < 8; ++q) { u[q] = lo[q]; u[q + 8] = hi[q]; }
      bf[ni] = __builtin_bit_cast(v16bf, u);
    }
    #pragma unroll
    for (int mi = 0; mi < 2; ++mi)
      #pragma unroll
      for (int ni = 0; ni < 2; ++ni)
        acc[mi][ni] = __builtin_amdgcn_wmma_f32_16x16x32_bf16(
            false, a[mi], false, bf[ni], (short)0, acc[mi][ni], false, false);
  }

  const int ncol = lane & 15;
  #pragma unroll
  for (int mi = 0; mi < 2; ++mi) {
    #pragma unroll
    for (int ni = 0; ni < 2; ++ni) {
      const int n  = nbase + ni * 16 + ncol;
      const int mb = mbase + mi * 16 + halfsel * 8;
      #pragma unroll
      for (int r = 0; r < 8; ++r)
        t[(size_t)(mb + r) * Hdim + n] = acc[mi][ni][r];
    }
  }
}

// agg[dst] += t[src] * (dinv[src]*dinv[dst]); 128 threads/edge, 4 channels/thread
__global__ void edge_scatter(const int* __restrict__ src, const int* __restrict__ dst,
                             const float* __restrict__ dinv, const float* __restrict__ t,
                             float* __restrict__ agg) {
  int tid = blockIdx.x * blockDim.x + threadIdx.x;
  int e = tid >> 7;
  if (e >= ET_) return;
  int c = (tid & 127) * 4;
  int s = src[e], d = dst[e];
  float norm = dinv[s] * dinv[d];
  const float4 v = *(const float4*)(t + (size_t)s * Hdim + c);
  float* ap = agg + (size_t)d * Hdim + c;
  atomicAdd(ap + 0, v.x * norm);
  atomicAdd(ap + 1, v.y * norm);
  atomicAdd(ap + 2, v.z * norm);
  atomicAdd(ap + 3, v.w * norm);
}

__global__ void node_conv(const float* __restrict__ agg, const float* __restrict__ t,
                          const float* __restrict__ dinv, const float* __restrict__ bc,
                          float* __restrict__ h, unsigned short* __restrict__ hb) {
  int idx = blockIdx.x * blockDim.x + threadIdx.x;
  if (idx >= BN_ * Hdim) return;
  int i = idx >> 9;
  int c = idx & 511;
  float di = dinv[i];
  float v  = agg[idx] + t[idx] * (di * di) + bc[c];
  float nh = h[idx] + lrelu(v);
  h[idx]  = nh;
  hb[idx] = f2bf(nh);
}

__global__ __launch_bounds__(512) void pool_kernel(const float* __restrict__ h,
                                                   float* __restrict__ pooled) {
  const int CHUNKS = 40;                    // 625 nodes per block
  int b  = blockIdx.x / CHUNKS;
  int ch = blockIdx.x % CHUNKS;
  int c  = threadIdx.x;                     // channel 0..511 -> coalesced rows
  int n0 = ch * (Nn / CHUNKS);
  float acc = 0.f;
  const float* base = h + ((size_t)b * Nn + n0) * Hdim + c;
  for (int n = 0; n < Nn / CHUNKS; ++n) acc += base[(size_t)n * Hdim];
  atomicAdd(&pooled[b * Hdim + c], acc * (1.0f / Nn));
}

// out[s,l,b,o] = pooled[b,:] . W{g|b}[l,:,o] + b{g|b}[l,o]
__global__ void final_kernel(const float* __restrict__ pooled,
                             const float* __restrict__ Wg, const float* __restrict__ bg,
                             const float* __restrict__ Wb, const float* __restrict__ bb,
                             float* __restrict__ out) {
  int idx = blockIdx.x * blockDim.x + threadIdx.x;
  const int total = 2 * LH * Bg * OUTD;
  if (idx >= total) return;
  int o = idx & (OUTD - 1);
  int b = (idx >> 8) & 3;
  int l = (idx >> 10) % LH;
  int s = idx / (LH * Bg * OUTD);
  const float* W    = (s == 0) ? Wg : Wb;
  const float* bias = (s == 0) ? bg : bb;
  float acc = bias[l * OUTD + o];
  const float* pr = pooled + b * Hdim;
  const float* wc = W + (size_t)l * Hdim * OUTD + o;
  for (int k = 0; k < Hdim; ++k) acc += pr[k] * wc[(size_t)k * OUTD];
  out[idx] = acc;
}

extern "C" void kernel_launch(void* const* d_in, const int* in_sizes, int n_in,
                              void* d_out, int out_size, void* d_ws, size_t ws_size,
                              hipStream_t stream) {
  (void)in_sizes; (void)n_in; (void)out_size; (void)ws_size;
  const float* x   = (const float*)d_in[0];
  const int*   src = (const int*)d_in[1];
  const int*   dst = (const int*)d_in[2];
  const float* W1  = (const float*)d_in[3];
  const float* b1  = (const float*)d_in[4];
  const float* Wc  = (const float*)d_in[5];
  const float* bc  = (const float*)d_in[6];
  const float* Wg  = (const float*)d_in[7];
  const float* bg  = (const float*)d_in[8];
  const float* Wb  = (const float*)d_in[9];
  const float* bb  = (const float*)d_in[10];
  float* out = (float*)d_out;

  char* wsp = (char*)d_ws;
  auto take = [&](size_t bytes) -> char* {
    char* p = wsp;
    wsp += (bytes + 255) & ~(size_t)255;
    return p;
  };
  float* deg   = (float*)take((size_t)BN_ * 4);
  float* dinv  = (float*)take((size_t)BN_ * 4);
  float* sagg  = (float*)take((size_t)BN_ * 4);
  float* h     = (float*)take((size_t)BN_ * Hdim * 4);
  float* t     = (float*)take((size_t)BN_ * Hdim * 4);
  float* agg   = (float*)take((size_t)BN_ * Hdim * 4);
  unsigned short* hb = (unsigned short*)take((size_t)BN_ * Hdim * 2);
  unsigned short* Wp = (unsigned short*)take((size_t)Hdim * Hdim * 2);
  float* pooled = (float*)take((size_t)Bg * Hdim * 4);

  const int TB = 256;
  // degree + symmetric norm
  fill_f32<<<(BN_ + TB - 1) / TB, TB, 0, stream>>>(deg, 1.0f, BN_);   // +1 self-loop
  fill_f32<<<(BN_ + TB - 1) / TB, TB, 0, stream>>>(sagg, 0.0f, BN_);
  edge_deg<<<(ET_ + TB - 1) / TB, TB, 0, stream>>>(dst, deg);
  make_dinv<<<(BN_ + TB - 1) / TB, TB, 0, stream>>>(deg, dinv);

  // layer 1 (rank-1 weight -> scalar aggregation)
  edge_scalar<<<(ET_ + TB - 1) / TB, TB, 0, stream>>>(src, dst, x, dinv, sagg);
  node_layer1<<<(BN_ * Hdim + TB - 1) / TB, TB, 0, stream>>>(x, dinv, sagg, W1, b1, h, hb);

  // 3 GCN residual conv layers
  for (int layer = 0; layer < NCONV; ++layer) {
    pack_w<<<(262144 + TB - 1) / TB, TB, 0, stream>>>(Wc + (size_t)layer * Hdim * Hdim, Wp);
    fill_f32<<<(BN_ * Hdim + TB - 1) / TB, TB, 0, stream>>>(agg, 0.0f, BN_ * Hdim);
    gemm_bf16<<<(BN_ / 32) * (Hdim / 32) / 8, TB, 0, stream>>>(hb, Wp, t);
    edge_scatter<<<(ET_ * 128) / TB, TB, 0, stream>>>(src, dst, dinv, t, agg);
    node_conv<<<(BN_ * Hdim + TB - 1) / TB, TB, 0, stream>>>(agg, t, dinv,
                                                             bc + layer * Hdim, h, hb);
  }

  // global mean pool + dual linear heads
  fill_f32<<<(Bg * Hdim + TB - 1) / TB, TB, 0, stream>>>(pooled, 0.0f, Bg * Hdim);
  pool_kernel<<<Bg * 40, 512, 0, stream>>>(h, pooled);
  final_kernel<<<(2 * LH * Bg * OUTD + TB - 1) / TB, TB, 0, stream>>>(pooled, Wg, bg, Wb, bb, out);
}
